// HyperMonarchLinear_4612794876561
// MI455X (gfx1250) — matrix operand order repaired and checked by reading, not compiled, not run on hardware
//
#include <hip/hip_runtime.h>
#include <hip/hip_bf16.h>
#include <stdint.h>

// ---- CDNA5 WMMA types ----
typedef __attribute__((ext_vector_type(16))) __bf16 v16bf;
typedef __attribute__((ext_vector_type(8)))  float  v8f;

#define CDIM    256
#define NG      32          // n_in == n_out == 32
#define DM      1024
#define WAVES   4           // wave32 -> 128 threads
#define BLOCK   (WAVES * 32)
#define TOKW    16          // tokens per wave (WMMA M)
#define TOKB    (WAVES * TOKW)
#define CCHUNK  8           // c-steps per staging round (16KB bf16)

union BF16Vec { unsigned int u[8]; v16bf v; };

// hardware packed f32->bf16 (RNE): dst.lo = cvt(lo), dst.hi = cvt(hi)
__device__ __forceinline__ unsigned int pk2(float lo, float hi) {
  unsigned int d;
  asm("v_cvt_pk_bf16_f32 %0, %1, %2" : "=v"(d) : "v"(lo), "v"(hi));
  return d;
}
__device__ __forceinline__ unsigned int bfsplat(float f) { return pk2(f, f); }
__device__ __forceinline__ unsigned short bf1(float f) { return (unsigned short)pk2(f, 0.0f); }

// packed bf16 multiply (VOP3P) — per-token cf scaling of the A operand
__device__ __forceinline__ unsigned int pkmul_bf16(unsigned int a, unsigned int b) {
  unsigned int d;
  asm("v_pk_mul_bf16 %0, %1, %2" : "=v"(d) : "v"(a), "v"(b));
  return d;
}

// async 16B global -> LDS copy (tracked by ASYNCcnt)
__device__ __forceinline__ void async_cp16(unsigned lds_off, const void* gsrc) {
  asm volatile("global_load_async_to_lds_b128 %0, %1, off"
               :: "v"(lds_off), "v"((unsigned long long)(uintptr_t)gsrc) : "memory");
}
__device__ __forceinline__ void wait_async0() {
  asm volatile("s_wait_asynccnt 0x0" ::: "memory");
}

// A-operand K index for v16bf element j (contiguous order, matches B packing):
// lanes 0-15: K0..15 ; lanes 16-31: K16..31
__device__ __forceinline__ int a_kof(int j, int lh) {
  return j + 16 * lh;
}

template <bool REUSE_A>
__device__ __forceinline__ v8f wmma_bf16(const BF16Vec& a, const BF16Vec& b, v8f c) {
  return __builtin_amdgcn_wmma_f32_16x16x32_bf16(false, a.v, false, b.v, (short)0, c,
                                                 REUSE_A, false);
}

// ---------------------------------------------------------------------------
// Preprocess: W[c, g, i, o] f32 -> bf16 in B-operand lane layout, reordered to
// [g][c][block of 1024] so a CCHUNK of c's is one contiguous 16KB stream.
// One thread per (i, i+1) pair -> one v_cvt_pk_bf16_f32 + one dword store.
// ---------------------------------------------------------------------------
__global__ __launch_bounds__(256)
void prep_w_kernel(const float* __restrict__ W, unsigned short* __restrict__ dst) {
  int q = blockIdx.x * blockDim.x + threadIdx.x;   // pair id: 0 .. NG*CDIM*512-1
  int cg = q >> 9;                                 // c*NG + g
  int r  = q & 511;
  int ip = r >> 5, o = r & 31, i = ip << 1;
  int c  = cg >> 5, g = cg & 31;
  float a = W[(size_t)cg * 1024 + (size_t)i * 32 + o];
  float b = W[(size_t)cg * 1024 + (size_t)(i + 1) * 32 + o];
  int l2  = (o & 15) | (i & 16);
  int pe2 = ((o >> 4) * 32 + l2) * 8 + (ip & 7);   // dword slot within block
  ((unsigned int*)dst)[((size_t)g * CDIM + c) * 512 + pe2] = pk2(a, b);
}

__global__ __launch_bounds__(256)
void prep_b_kernel(const float* __restrict__ bias, unsigned short* __restrict__ dst) {
  int q = blockIdx.x * blockDim.x + threadIdx.x;   // 0 .. NG*512-1
  int g = q >> 9;
  int r = q & 511;
  int ip = r >> 5, o = r & 31, i = ip << 1;
  float a = bias[g * 1024 + i * 32 + o];
  float b = bias[g * 1024 + (i + 1) * 32 + o];
  int l2  = (o & 15) | (i & 16);
  int pe2 = ((o >> 4) * 32 + l2) * 8 + (ip & 7);
  ((unsigned int*)dst)[g * 512 + pe2] = pk2(a, b);
}

// ---------------------------------------------------------------------------
// Shared per-stage helpers
// ---------------------------------------------------------------------------
__device__ __forceinline__ void load_cf_splats(const float* cf, int tok, int cb,
                                               unsigned int cfp[8]) {
  const float4 c0 = *(const float4*)(cf + (size_t)tok * CDIM + cb);
  const float4 c1 = *(const float4*)(cf + (size_t)tok * CDIM + cb + 4);
  cfp[0] = bfsplat(c0.x); cfp[1] = bfsplat(c0.y);
  cfp[2] = bfsplat(c0.z); cfp[3] = bfsplat(c0.w);
  cfp[4] = bfsplat(c1.x); cfp[5] = bfsplat(c1.y);
  cfp[6] = bfsplat(c1.z); cfp[7] = bfsplat(c1.w);
}

// ---------------------------------------------------------------------------
// FAST kernel: W already bf16 + permuted in ws; staging = async copy to LDS,
// double-buffered against WMMA.
// ---------------------------------------------------------------------------
__global__ __launch_bounds__(BLOCK)
void hm_fast_kernel(const float* __restrict__ x, const float* __restrict__ cf,
                    const unsigned short* __restrict__ wsW,
                    const unsigned short* __restrict__ wsB,
                    float* __restrict__ out)
{
  __shared__ unsigned int sWu[2][CCHUNK * 512];        // 2 x 16KB bf16 W chunks
  __shared__ unsigned int sBu[512];                    // 2KB bias block
  __shared__ unsigned int sSu[WAVES * TOKW * NG * NG / 2]; // 128KB stage-1 out
  unsigned short* sS = (unsigned short*)sSu;

  const int tid  = threadIdx.x;
  const int wave = tid >> 5;
  const int lane = tid & 31;
  const int lh   = lane >> 4;
  const int lm   = lane & 15;
  const int tok0 = blockIdx.x * TOKB + wave * TOKW;

  const unsigned ldsW[2] = { (unsigned)(uintptr_t)&sWu[0][0],
                             (unsigned)(uintptr_t)&sWu[1][0] };
  const unsigned ldsB = (unsigned)(uintptr_t)&sBu[0];

  for (int stage = 0; stage < 2; ++stage) {
    const unsigned short* Wb = wsW + (size_t)stage * NG * CDIM * 1024;
    const unsigned short* Bb = wsB + (size_t)stage * NG * 1024;

    for (int g = 0; g < NG; ++g) {
      // ---- A base tile (16 tokens x 32 k) as packed bf16 ----
      BF16Vec xp;
      if (stage == 0) {
        const float* xr = x + (size_t)(tok0 + lm) * DM + g * NG;
#pragma unroll
        for (int v = 0; v < 8; ++v) {
          int k0 = a_kof(2 * v, lh);
          xp.u[v] = pk2(xr[k0], xr[k0 + 1]);
        }
      } else {
        const unsigned short* sr = sS + (((wave * TOKW + lm) * NG + g) * NG);
#pragma unroll
        for (int v = 0; v < 8; ++v) xp.u[v] = *(const unsigned int*)(sr + a_kof(2 * v, lh));
      }

      v8f acc0 = {}; v8f acc1 = {};

      const unsigned char* gW = (const unsigned char*)(Wb + (size_t)g * CDIM * 1024);

      __syncthreads();   // buffers free of previous g's readers
      // prefetch chunk 0 -> buf 0 (8 x 16B per thread = 16KB per block)
#pragma unroll
      for (int k = 0; k < 8; ++k)
        async_cp16(ldsW[0] + tid * 16 + k * 2048, gW + tid * 16 + k * 2048);

      for (int cb = 0; cb < CDIM; cb += CCHUNK) {
        const int cur = (cb >> 3) & 1;
        wait_async0();      // current buffer resident
        __syncthreads();    // all waves ready; other buffer's readers done
        if (cb + CCHUNK < CDIM) {
          const unsigned char* src = gW + (size_t)(cb + CCHUNK) * 2048;
#pragma unroll
          for (int k = 0; k < 8; ++k)
            async_cp16(ldsW[cur ^ 1] + tid * 16 + k * 2048, src + tid * 16 + k * 2048);
        }

        unsigned int cfp[8];
        load_cf_splats(cf, tok0 + lm, cb, cfp);

#pragma unroll
        for (int cc = 0; cc < CCHUNK; ++cc) {
          BF16Vec a;
#pragma unroll
          for (int v = 0; v < 8; ++v) a.u[v] = pkmul_bf16(xp.u[v], cfp[cc]);
          BF16Vec bv0, bv1;
          const unsigned int* bp0 = &sWu[cur][(((cc << 1) | 0) * 32 + lane) * 8];
          const unsigned int* bp1 = &sWu[cur][(((cc << 1) | 1) * 32 + lane) * 8];
#pragma unroll
          for (int v = 0; v < 8; ++v) { bv0.u[v] = bp0[v]; bv1.u[v] = bp1[v]; }
          acc0 = wmma_bf16<false>(a, bv0, acc0);
          acc1 = wmma_bf16<true>(a, bv1, acc1);   // reuse-A hint: identical op
        }
      }

      // ---- bias as one extra K-step with scale 1.0 ----
      __syncthreads();                 // previous bias readers done
      async_cp16(ldsB + tid * 16, (const unsigned char*)(Bb + g * 1024) + tid * 16);
      wait_async0();
      __syncthreads();
      {
        BF16Vec bv0, bv1;
        const unsigned int* bp0 = &sBu[(0 * 32 + lane) * 8];
        const unsigned int* bp1 = &sBu[(1 * 32 + lane) * 8];
#pragma unroll
        for (int v = 0; v < 8; ++v) { bv0.u[v] = bp0[v]; bv1.u[v] = bp1[v]; }
        acc0 = wmma_bf16<false>(xp, bv0, acc0);
        acc1 = wmma_bf16<true>(xp, bv1, acc1);
      }

      // ---- writeback ----
      if (stage == 0) {
#pragma unroll
        for (int r = 0; r < 8; ++r) {
          int t = r + (lh << 3);
          sS[((wave * TOKW + t) * NG + lm) * NG + g]        = bf1(acc0[r]);
          sS[((wave * TOKW + t) * NG + (lm + 16)) * NG + g] = bf1(acc1[r]);
        }
      } else {
#pragma unroll
        for (int r = 0; r < 8; ++r) {
          int t = r + (lh << 3);
          out[(size_t)(tok0 + t) * DM + g * NG + lm]      = acc0[r];
          out[(size_t)(tok0 + t) * DM + g * NG + 16 + lm] = acc1[r];
        }
      }
    }
    __syncthreads();   // stage-1 S fully written before stage-2 A reads
  }
}

// ---------------------------------------------------------------------------
// FALLBACK kernel (ws too small): convert/permute W blocks in-kernel.
// ---------------------------------------------------------------------------
__global__ __launch_bounds__(BLOCK)
void hm_fallback_kernel(const float* __restrict__ x, const float* __restrict__ cf,
                        const float* __restrict__ W1, const float* __restrict__ b1,
                        const float* __restrict__ W2, const float* __restrict__ b2,
                        float* __restrict__ out)
{
  __shared__ unsigned int sWu[CCHUNK * 512];               // 16KB
  __shared__ unsigned int sSu[WAVES * TOKW * NG * NG / 2]; // 128KB
  unsigned short* sS = (unsigned short*)sSu;

  const int tid  = threadIdx.x;
  const int wave = tid >> 5;
  const int lane = tid & 31;
  const int lh   = lane >> 4;
  const int lm   = lane & 15;
  const int tok0 = blockIdx.x * TOKB + wave * TOKW;

  for (int stage = 0; stage < 2; ++stage) {
    const float* W    = stage ? W2 : W1;
    const float* bias = stage ? b2 : b1;

    for (int g = 0; g < NG; ++g) {
      BF16Vec xp;
      if (stage == 0) {
        const float* xr = x + (size_t)(tok0 + lm) * DM + g * NG;
#pragma unroll
        for (int v = 0; v < 8; ++v) {
          int k0 = a_kof(2 * v, lh);
          xp.u[v] = pk2(xr[k0], xr[k0 + 1]);
        }
      } else {
        const unsigned short* sr = sS + (((wave * TOKW + lm) * NG + g) * NG);
#pragma unroll
        for (int v = 0; v < 8; ++v) xp.u[v] = *(const unsigned int*)(sr + a_kof(2 * v, lh));
      }

      v8f acc0 = {}; v8f acc1 = {};

      for (int cb = 0; cb < CDIM; cb += CCHUNK) {
        __syncthreads();
        // pairwise convert+permute: 1 cvt_pk + 1 dword LDS store per 2 elems
        for (int q = tid; q < CCHUNK * 512; q += BLOCK) {
          int cc = q >> 9, r = q & 511;
          int ip = r >> 5, o = r & 31, i = ip << 1;
          const float* src = W + (size_t)(cb + cc) * (NG * NG * NG) + (size_t)g * 1024;
          float a = src[i * 32 + o], b = src[(i + 1) * 32 + o];
          int l2 = (o & 15) | (i & 16);
          sWu[(((cc << 1) | (o >> 4)) * 32 + l2) * 8 + (ip & 7)] = pk2(a, b);
        }
        __syncthreads();

        unsigned int cfp[8];
        load_cf_splats(cf, tok0 + lm, cb, cfp);

#pragma unroll
        for (int cc = 0; cc < CCHUNK; ++cc) {
          BF16Vec a;
#pragma unroll
          for (int v = 0; v < 8; ++v) a.u[v] = pkmul_bf16(xp.u[v], cfp[cc]);
          BF16Vec bv0, bv1;
          const unsigned int* bp0 = &sWu[(((cc << 1) | 0) * 32 + lane) * 8];
          const unsigned int* bp1 = &sWu[(((cc << 1) | 1) * 32 + lane) * 8];
#pragma unroll
          for (int v = 0; v < 8; ++v) { bv0.u[v] = bp0[v]; bv1.u[v] = bp1[v]; }
          acc0 = wmma_bf16<false>(a, bv0, acc0);
          acc1 = wmma_bf16<true>(a, bv1, acc1);
        }
      }

      __syncthreads();
      for (int q = tid; q < 512; q += BLOCK) {
        int ip = q >> 5, o = q & 31, i = ip << 1;
        float a = bias[g * 1024 + i * 32 + o], b = bias[g * 1024 + (i + 1) * 32 + o];
        int l2 = (o & 15) | (i & 16);
        sWu[((o >> 4) * 32 + l2) * 8 + (ip & 7)] = pk2(a, b);
      }
      __syncthreads();
      {
        BF16Vec bv0, bv1;
        const unsigned int* bp0 = &sWu[(0 * 32 + lane) * 8];
        const unsigned int* bp1 = &sWu[(1 * 32 + lane) * 8];
#pragma unroll
        for (int v = 0; v < 8; ++v) { bv0.u[v] = bp0[v]; bv1.u[v] = bp1[v]; }
        acc0 = wmma_bf16<false>(xp, bv0, acc0);
        acc1 = wmma_bf16<true>(xp, bv1, acc1);
      }

      if (stage == 0) {
#pragma unroll
        for (int r = 0; r < 8; ++r) {
          int t = r + (lh << 3);
          sS[((wave * TOKW + t) * NG + lm) * NG + g]        = bf1(acc0[r]);
          sS[((wave * TOKW + t) * NG + (lm + 16)) * NG + g] = bf1(acc1[r]);
        }
      } else {
#pragma unroll
        for (int r = 0; r < 8; ++r) {
          int t = r + (lh << 3);
          out[(size_t)(tok0 + t) * DM + g * NG + lm]      = acc0[r];
          out[(size_t)(tok0 + t) * DM + g * NG + 16 + lm] = acc1[r];
        }
      }
    }
    __syncthreads();
  }
}

extern "C" void kernel_launch(void* const* d_in, const int* in_sizes, int n_in,
                              void* d_out, int out_size, void* d_ws, size_t ws_size,
                              hipStream_t stream) {
  const float* x  = (const float*)d_in[0];
  const float* cf = (const float*)d_in[1];
  const float* W1 = (const float*)d_in[2];
  const float* b1 = (const float*)d_in[3];
  const float* W2 = (const float*)d_in[4];
  const float* b2 = (const float*)d_in[5];
  float* out = (float*)d_out;

  const int tokens = in_sizes[0] / DM;            // 4096
  const int grid   = (tokens + TOKB - 1) / TOKB;  // 64 blocks of 4 waves

  const size_t NW = (size_t)NG * CDIM * 1024;     // bf16 elems per stage (W)
  const size_t NB = (size_t)NG * 1024;            // bf16 elems per stage (bias)
  const size_t WS_NEED = (2 * NW + 2 * NB) * sizeof(unsigned short); // ~33.7MB

  if (ws_size >= WS_NEED) {
    unsigned short* wsW = (unsigned short*)d_ws;
    unsigned short* wsB = wsW + 2 * NW;
    const int wPairs = (int)(NW / 2);             // 4,194,304 per stage
    const int bPairs = (int)(NB / 2);             // 16,384 per stage
    prep_w_kernel<<<wPairs / 256, 256, 0, stream>>>(W1, wsW);
    prep_w_kernel<<<wPairs / 256, 256, 0, stream>>>(W2, wsW + NW);
    prep_b_kernel<<<bPairs / 256, 256, 0, stream>>>(b1, wsB);
    prep_b_kernel<<<bPairs / 256, 256, 0, stream>>>(b2, wsB + NB);
    hm_fast_kernel<<<grid, BLOCK, 0, stream>>>(x, cf, wsW, wsB, out);
  } else {
    hm_fallback_kernel<<<grid, BLOCK, 0, stream>>>(x, cf, W1, b1, W2, b2, out);
  }
}
